// GRU_55740085567602
// MI455X (gfx1250) — matrix operand order repaired and checked by reading, not compiled, not run on hardware
//
#include <hip/hip_runtime.h>
#include <hip/hip_bf16.h>

// GRU+LN for MI455X (gfx1250), wave32, WMMA bf16 path.
//
// Phase A: precomp[t,b,0:2H]   = x(b,t,:) @ Wg_x^T + bg_x   (gates input term)
//          precomp[t,b,2H:3H]  = x(b,t,:) @ Wx_x^T + bx_x   (candidate input term)
//          -> one (32768 x 3072 x 512) bf16 WMMA GEMM.
// Phase B: 512 sequential steps, each:
//          step_gemm : Gpre[b,0:2H]  = precomp[t,b,0:2H] + bg_h + h @ Wg_h^T
//                      Gpre[b,2H:3H] = bx_h + h @ Wx_h^T
//          step_point: LN(2H)->sigmoid gates (r,u); z = Gpre_xh*r + precomp_xx;
//                      LN(H); hn = u*h + (1-u)*tanh(.); update h (f32+bf16), write out.
//
// WMMA operand layouts used here (same contiguous k order for A and B):
//   A 16x32 bf16: lane<16 -> row m0+lane, K = k0+0..15 (contiguous)
//                 lane>=16 -> row m0+lane-16, K = k0+16..31 (contiguous)
//   B 32x16 bf16: lane<16 -> col n0+lane, K = k0+0..15 (contiguous)
//                 lane>=16 -> col n0+lane-16, K = k0+16..31 (contiguous)
//
// Workspace usage (~426 MiB): precomp f32 (384 MiB), bf16 x (32 MiB),
// bf16 packed weights (9 MiB), Gpre / h state (<2 MiB).

#define B_  64
#define T_  512
#define I_  512
#define H_  1024
#define G2  (2*H_)   // 2048
#define G3  (3*H_)   // 3072
#define EPSV 0.001f

typedef __attribute__((ext_vector_type(16))) __bf16 v16bf;
typedef __attribute__((ext_vector_type(8)))  float  v8f;

static __device__ __forceinline__ unsigned short f2bf(float x) {
    unsigned int u = __float_as_uint(x);
    unsigned int r = (u + 0x7FFFu + ((u >> 16) & 1u)) >> 16;  // round-to-nearest-even
    return (unsigned short)r;
}

union ABu { v16bf v; uint4 q[2]; };

// A-operand lane slice: p = &A[row][k0 + 16*hi]
// elements 0..15 <- K = k0 + 16*hi + 0..15 (contiguous)
static __device__ __forceinline__ v16bf load16A(const unsigned short* p) {
    ABu u;
    u.q[0] = *(const uint4*)(p);
    u.q[1] = *(const uint4*)(p + 8);
    return u.v;
}

// B-operand lane slice: p = &W[col][k0 + 16*hi]  (W stored N-major, row = output col)
// elements 0..15 <- K = k0 + 16*hi + 0..15 (contiguous)
static __device__ __forceinline__ v16bf load16B(const unsigned short* p) {
    ABu u;
    u.q[0] = *(const uint4*)(p);
    u.q[1] = *(const uint4*)(p + 8);
    return u.v;
}

// ---------------------------------------------------------------------------
// Phase A GEMM: precomp(T*B, 3H) = Xbf(T*B, I) @ WxPack(3H, I)^T + bias
// grid = (3072/256, 32768/16), block = 128 (4 waves), wave tile = 16x64.
// ---------------------------------------------------------------------------
__global__ void pre_gemm(const unsigned short* __restrict__ xbf,   // (B,T,I) bf16
                         const unsigned short* __restrict__ Wx,    // (3H,I) bf16
                         const float* __restrict__ bg_x,           // (2H)
                         const float* __restrict__ bx_x,           // (H)
                         float* __restrict__ precomp)              // (T*B, 3H)
{
    const int lane = threadIdx.x & 31;
    const int wave = threadIdx.x >> 5;
    const int hi   = lane >> 4;
    const int l15  = lane & 15;
    const int m0   = blockIdx.y * 16;
    const int n0   = blockIdx.x * 256 + wave * 64;

    // A row for this lane: m = t*B + b, read from x[b, t, :]
    const int mrowA = m0 + l15;
    const int t = mrowA >> 6;      // /64
    const int b = mrowA & 63;      // %64
    const unsigned short* arow = xbf + (size_t)b * (T_ * I_) + (size_t)t * I_;

    v8f acc[4] = {};
    for (int k0 = 0; k0 < I_; k0 += 32) {
        v16bf a = load16A(arow + k0 + hi * 16);
#pragma unroll
        for (int j = 0; j < 4; ++j) {
            v16bf bb = load16B(Wx + (size_t)(n0 + 16 * j + l15) * I_ + k0 + hi * 16);
            acc[j] = __builtin_amdgcn_wmma_f32_16x16x32_bf16(
                false, a, false, bb, (short)0, acc[j], false, false);
        }
    }
#pragma unroll
    for (int j = 0; j < 4; ++j) {
        const int n = n0 + 16 * j + l15;
        const float bias = (n < G2) ? bg_x[n] : bx_x[n - G2];
#pragma unroll
        for (int r = 0; r < 8; ++r) {
            const int m = m0 + r + 8 * hi;
            precomp[(size_t)m * G3 + n] = acc[j][r] + bias;
        }
    }
}

// ---------------------------------------------------------------------------
// Per-step GEMM: Gpre(B,3H) = h(B,H) @ WhPack(3H,H)^T + terms
// grid = (3072/256, 64/16) = (12,4), block = 128.
// ---------------------------------------------------------------------------
__global__ void step_gemm(const unsigned short* __restrict__ hbf,  // (B,H) bf16
                          const unsigned short* __restrict__ Wh,   // (3H,H) bf16
                          const float* __restrict__ precomp_t,     // (B,3H) slice at t
                          const float* __restrict__ bg_h,          // (2H)
                          const float* __restrict__ bx_h,          // (H)
                          float* __restrict__ Gpre)                // (B,3H)
{
    const int lane = threadIdx.x & 31;
    const int wave = threadIdx.x >> 5;
    const int hi   = lane >> 4;
    const int l15  = lane & 15;
    const int m0   = blockIdx.y * 16;
    const int n0   = blockIdx.x * 256 + wave * 64;

    const unsigned short* arow = hbf + (size_t)(m0 + l15) * H_;

    v8f acc[4] = {};
    for (int k0 = 0; k0 < H_; k0 += 32) {
        v16bf a = load16A(arow + k0 + hi * 16);
#pragma unroll
        for (int j = 0; j < 4; ++j) {
            v16bf bb = load16B(Wh + (size_t)(n0 + 16 * j + l15) * H_ + k0 + hi * 16);
            acc[j] = __builtin_amdgcn_wmma_f32_16x16x32_bf16(
                false, a, false, bb, (short)0, acc[j], false, false);
        }
    }
#pragma unroll
    for (int j = 0; j < 4; ++j) {
        const int n = n0 + 16 * j + l15;
#pragma unroll
        for (int r = 0; r < 8; ++r) {
            const int m = m0 + r + 8 * hi;   // batch row b
            float v = acc[j][r];
            if (n < G2) v += precomp_t[(size_t)m * G3 + n] + bg_h[n];
            else        v += bx_h[n - G2];
            Gpre[(size_t)m * G3 + n] = v;
        }
    }
}

// ---------------------------------------------------------------------------
// Per-step pointwise: LN over gates (2H), sigmoid, LN over candidate (H),
// h update + output store. One workgroup per batch row. block = 256 (8 waves).
// ---------------------------------------------------------------------------
__global__ void step_point(const float* __restrict__ Gpre,       // (B,3H)
                           const float* __restrict__ precomp_t,  // (B,3H)
                           const float* __restrict__ a_g, const float* __restrict__ b_g,
                           const float* __restrict__ a_x, const float* __restrict__ b_x,
                           float* __restrict__ hf,               // (B,H) state f32
                           unsigned short* __restrict__ hbf,     // (B,H) state bf16
                           float* __restrict__ out,              // (B,T,H)
                           int t)
{
    __shared__ float sr[H_];
    __shared__ float su[H_];
    __shared__ float redA[8];
    __shared__ float redB[8];

    const int b   = blockIdx.x;
    const int tid = threadIdx.x;
    const int wave = tid >> 5;
    const int lane = tid & 31;

    // ---- phase 1: LN over 2H gate pre-activations ----
    float gv[8];
    float s = 0.f, sq = 0.f;
#pragma unroll
    for (int i = 0; i < 8; ++i) {
        const int j = tid + i * 256;
        const float g = Gpre[(size_t)b * G3 + j];
        gv[i] = g; s += g; sq += g * g;
    }
#pragma unroll
    for (int off = 16; off > 0; off >>= 1) {
        s  += __shfl_down(s,  off, 32);
        sq += __shfl_down(sq, off, 32);
    }
    if (lane == 0) { redA[wave] = s; redB[wave] = sq; }
    __syncthreads();
    if (tid == 0) {
        float a = 0.f, c = 0.f;
        for (int w = 0; w < 8; ++w) { a += redA[w]; c += redB[w]; }
        redA[0] = a; redB[0] = c;
    }
    __syncthreads();
    s = redA[0]; sq = redB[0];

    const float mu  = s / (float)G2;
    const float var = (sq - (float)G2 * mu * mu) / (float)(G2 - 1);
    const float inv = 1.0f / (sqrtf(var) + EPSV);

    __syncthreads();  // protect redA/redB reuse
#pragma unroll
    for (int i = 0; i < 8; ++i) {
        const int j = tid + i * 256;
        const float z = (gv[i] - mu) * inv * a_g[j] + b_g[j];
        const float gate = 1.0f / (1.0f + __expf(-z));
        if (j < H_) sr[j] = gate; else su[j - H_] = gate;
    }
    __syncthreads();

    // ---- phase 2: candidate preact LN over H ----
    float zv[4];
    s = 0.f; sq = 0.f;
#pragma unroll
    for (int i = 0; i < 4; ++i) {
        const int j = tid + i * 256;
        const float z = Gpre[(size_t)b * G3 + G2 + j] * sr[j] +
                        precomp_t[(size_t)b * G3 + G2 + j];
        zv[i] = z; s += z; sq += z * z;
    }
#pragma unroll
    for (int off = 16; off > 0; off >>= 1) {
        s  += __shfl_down(s,  off, 32);
        sq += __shfl_down(sq, off, 32);
    }
    if (lane == 0) { redA[wave] = s; redB[wave] = sq; }
    __syncthreads();
    if (tid == 0) {
        float a = 0.f, c = 0.f;
        for (int w = 0; w < 8; ++w) { a += redA[w]; c += redB[w]; }
        redA[0] = a; redB[0] = c;
    }
    __syncthreads();
    s = redA[0]; sq = redB[0];

    const float mu2  = s / (float)H_;
    const float var2 = (sq - (float)H_ * mu2 * mu2) / (float)(H_ - 1);
    const float inv2 = 1.0f / (sqrtf(var2) + EPSV);

#pragma unroll
    for (int i = 0; i < 4; ++i) {
        const int j = tid + i * 256;
        const float pre = (zv[i] - mu2) * inv2 * a_x[j] + b_x[j];
        const float h   = hf[(size_t)b * H_ + j];
        const float u   = su[j];
        const float hn  = u * h + (1.0f - u) * tanhf(pre);
        hf[(size_t)b * H_ + j]  = hn;
        hbf[(size_t)b * H_ + j] = f2bf(hn);
        out[(size_t)b * (T_ * H_) + (size_t)t * H_ + j] = hn;
    }
}

// ---------------------------------------------------------------------------
// Utility kernels
// ---------------------------------------------------------------------------
__global__ void cvt_f32_bf16(const float* __restrict__ src,
                             unsigned short* __restrict__ dst, int n) {
    const int i = blockIdx.x * 256 + threadIdx.x;
    if (i < n) dst[i] = f2bf(src[i]);
}

__global__ void init_h(const float* __restrict__ h0,
                       float* __restrict__ hf,
                       unsigned short* __restrict__ hbf, int n) {
    const int i = blockIdx.x * 256 + threadIdx.x;
    if (i < n) { const float v = h0[i]; hf[i] = v; hbf[i] = f2bf(v); }
}

__global__ void copy_f32(const float* __restrict__ src,
                         float* __restrict__ dst, int n) {
    const int i = blockIdx.x * 256 + threadIdx.x;
    if (i < n) dst[i] = src[i];
}

// ---------------------------------------------------------------------------
extern "C" void kernel_launch(void* const* d_in, const int* in_sizes, int n_in,
                              void* d_out, int out_size, void* d_ws, size_t ws_size,
                              hipStream_t stream) {
    const float* inp    = (const float*)d_in[0];   // (B,T,I)
    const float* hidden = (const float*)d_in[1];   // (1,B,H)
    const float* Wg_x   = (const float*)d_in[2];   // (2H,I)
    const float* bg_x   = (const float*)d_in[3];   // (2H)
    const float* Wg_h   = (const float*)d_in[4];   // (2H,H)
    const float* bg_h   = (const float*)d_in[5];   // (2H)
    const float* Wx_x   = (const float*)d_in[6];   // (H,I)
    const float* bx_x   = (const float*)d_in[7];   // (H)
    const float* Wx_h   = (const float*)d_in[8];   // (H,H)
    const float* bx_h   = (const float*)d_in[9];   // (H)
    const float* a_g    = (const float*)d_in[10];
    const float* b_g    = (const float*)d_in[11];
    const float* a_x    = (const float*)d_in[12];
    const float* b_x    = (const float*)d_in[13];
    float* out = (float*)d_out;                    // (B,T,H) then (B,H)

    // Workspace carve-up (256B aligned)
    char* ws = (char*)d_ws;
    size_t off = 0;
    auto carve = [&](size_t bytes) -> char* {
        char* p = ws + off;
        off = (off + bytes + 255) & ~(size_t)255;
        return p;
    };
    float*          precomp = (float*)         carve((size_t)T_ * B_ * G3 * 4); // 384 MiB
    unsigned short* xbf     = (unsigned short*)carve((size_t)B_ * T_ * I_ * 2); // 32 MiB
    unsigned short* WxP     = (unsigned short*)carve((size_t)G3 * I_ * 2);      // 3 MiB
    unsigned short* WhP     = (unsigned short*)carve((size_t)G3 * H_ * 2);      // 6 MiB
    float*          Gpre    = (float*)         carve((size_t)B_ * G3 * 4);
    float*          hf      = (float*)         carve((size_t)B_ * H_ * 4);
    unsigned short* hbf     = (unsigned short*)carve((size_t)B_ * H_ * 2);
    (void)ws_size;

    auto blocks = [](int n) { return (n + 255) / 256; };

    // Pack / convert operands to bf16
    cvt_f32_bf16<<<blocks(G2 * I_), 256, 0, stream>>>(Wg_x, WxP, G2 * I_);
    cvt_f32_bf16<<<blocks(H_ * I_), 256, 0, stream>>>(Wx_x, WxP + (size_t)G2 * I_, H_ * I_);
    cvt_f32_bf16<<<blocks(G2 * H_), 256, 0, stream>>>(Wg_h, WhP, G2 * H_);
    cvt_f32_bf16<<<blocks(H_ * H_), 256, 0, stream>>>(Wx_h, WhP + (size_t)G2 * H_, H_ * H_);
    cvt_f32_bf16<<<blocks(B_ * T_ * I_), 256, 0, stream>>>(inp, xbf, B_ * T_ * I_);
    init_h<<<blocks(B_ * H_), 256, 0, stream>>>(hidden, hf, hbf, B_ * H_);

    // Phase A: time-parallel input projections (WMMA GEMM)
    pre_gemm<<<dim3(G3 / 256, (T_ * B_) / 16), 128, 0, stream>>>(
        xbf, WxP, bg_x, bx_x, precomp);

    // Phase B: sequential recurrence
    for (int t = 0; t < T_; ++t) {
        const float* precomp_t = precomp + (size_t)t * B_ * G3;
        step_gemm<<<dim3(G3 / 256, B_ / 16), 128, 0, stream>>>(
            hbf, WhP, precomp_t, bg_h, bx_h, Gpre);
        step_point<<<B_, 256, 0, stream>>>(
            Gpre, precomp_t, a_g, b_g, a_x, b_x, hf, hbf, out, t);
    }

    // h_last -> tail of d_out
    copy_f32<<<blocks(B_ * H_), 256, 0, stream>>>(hf, out + (size_t)B_ * T_ * H_, B_ * H_);
    (void)in_sizes; (void)n_in; (void)out_size;
}
